// TrustGNN_86122684219560
// MI455X (gfx1250) — compile-verified, matrix operand order checked by reading, not run on hardware
//
#include <hip/hip_runtime.h>
#include <hip/hip_bf16.h>

#define N_NODES 50000
#define NEDGE   800000
#define EMB     384
#define NODE_F  (EMB + 4)        // 388
#define EDGE_F  (EMB + 1)        // 385
#define HEADS   2
#define HID     32
#define HC      (HEADS * HID)    // 64
#define EDGE_IN (2 * HC + EMB)   // 512
#define E2      (NEDGE + N_NODES) // 850000 (edges + self loops)
#define NEG_SLOPE 0.2f

typedef __attribute__((ext_vector_type(16))) __bf16 v16bf;
typedef __attribute__((ext_vector_type(8)))  float  v8f;

// K index inside a 32-wide chunk for vector element e, lane-half h
// (CDNA5 ISA 7.12.2: 16-bit A 16x32 / B 32x16 fragment layout)
__device__ __forceinline__ int kmap(int e, int h) {
    return ((e >> 3) << 4) + (h << 3) + (e & 7);
}

__device__ __forceinline__ v8f wmma_bf16(v16bf a, v16bf b, v8f c) {
    return __builtin_amdgcn_wmma_f32_16x16x32_bf16(false, a, false, b, (short)0, c, false, false);
}

// A fragment: 16B-aligned rows -> four float4 loads (two 8-contiguous groups)
__device__ __forceinline__ v16bf load_a_f4(const float* __restrict__ row, int kb, int h) {
    const float4* p0 = (const float4*)(row + kb + 8 * h);
    const float4* p1 = (const float4*)(row + kb + 16 + 8 * h);
    float4 f0 = p0[0], f1 = p0[1], f2 = p1[0], f3 = p1[1];
    v16bf a;
    a[0] = (__bf16)f0.x; a[1] = (__bf16)f0.y; a[2]  = (__bf16)f0.z; a[3]  = (__bf16)f0.w;
    a[4] = (__bf16)f1.x; a[5] = (__bf16)f1.y; a[6]  = (__bf16)f1.z; a[7]  = (__bf16)f1.w;
    a[8] = (__bf16)f2.x; a[9] = (__bf16)f2.y; a[10] = (__bf16)f2.z; a[11] = (__bf16)f2.w;
    a[12] = (__bf16)f3.x; a[13] = (__bf16)f3.y; a[14] = (__bf16)f3.z; a[15] = (__bf16)f3.w;
    return a;
}

// A fragment: scalar b32 loads, no predicates (full chunk)
__device__ __forceinline__ v16bf load_a_s(const float* __restrict__ row, int kb, int h) {
    v16bf a;
#pragma unroll
    for (int e = 0; e < 16; ++e) a[e] = (__bf16)row[kb + kmap(e, h)];
    return a;
}

// A fragment: tail chunk, branchless clamp + select (no exec-mask divergence)
__device__ __forceinline__ v16bf load_a_tail(const float* __restrict__ row, int kb, int h, int kdim) {
    v16bf a;
#pragma unroll
    for (int e = 0; e < 16; ++e) {
        int k = kb + kmap(e, h);
        float f = row[k < kdim ? k : 0];
        a[e] = (__bf16)(k < kdim ? f : 0.0f);
    }
    return a;
}

// B fragment from pre-packed buffer: one contiguous 32B load
__device__ __forceinline__ v16bf load_b_pk(const __bf16* __restrict__ pk, int chunk, int t, int lane) {
    return ((const v16bf*)pk)[(chunk * 4 + t) * 32 + lane];
}

__device__ __forceinline__ int aug_src(const int* ei, int e) {
    return (e < NEDGE) ? ei[e] : (e - NEDGE);
}
__device__ __forceinline__ int aug_tgt(const int* ei, int e) {
    return (e < NEDGE) ? ei[NEDGE + e] : (e - NEDGE);
}

// order-preserving float<->uint for atomic max (0 == ordered minimum)
__device__ __forceinline__ unsigned f2ord(float f) {
    unsigned u = __float_as_uint(f);
    return (u & 0x80000000u) ? ~u : (u | 0x80000000u);
}
__device__ __forceinline__ float ord2f(unsigned u) {
    return __uint_as_float((u & 0x80000000u) ? (u ^ 0x80000000u) : ~u);
}

// ---------------- kernels ----------------

__global__ void k_zero(float* p, size_t n) {
    size_t i = (size_t)blockIdx.x * blockDim.x + threadIdx.x;
    if (i < n) p[i] = 0.0f;
}

// pre-pack a (kdim x 64) f32 weight matrix into bf16 WMMA B fragments,
// zero-padded to nchunk*32 along K.  Layout: [chunk][t][lane][e] (16 bf16 = 32B)
__global__ void k_pack_w(const float* __restrict__ w, int kdim, int nchunk,
                         __bf16* __restrict__ pk) {
    int i = blockIdx.x * blockDim.x + threadIdx.x;
    int total = nchunk * 2048;            // 4 tiles * 32 lanes * 16 elems
    if (i >= total) return;
    int e = i & 15;
    int L = (i >> 4) & 31;
    int t = (i >> 9) & 3;
    int c = i >> 11;
    int k = c * 32 + kmap(e, L >> 4);
    int n = L & 15;
    pk[i] = (__bf16)((k < kdim) ? w[(size_t)k * HC + t * 16 + n] : 0.0f);
}

// segment-sum of edge_attr by target + degree count (one wave per edge)
__global__ void k_scatter_attr(const int* __restrict__ ei, const float* __restrict__ ea,
                               float* __restrict__ cnt, float* __restrict__ loop) {
    int lane = threadIdx.x & 31;
    int wv = blockIdx.x * (blockDim.x >> 5) + (threadIdx.x >> 5);
    if (wv >= NEDGE) return;
    int tg = ei[NEDGE + wv];
    const float* row = ea + (size_t)wv * EDGE_F;
    float* dst = loop + (size_t)tg * EDGE_F;
    for (int f = lane; f < EDGE_F; f += 32) atomicAdd(&dst[f], row[f]);
    if (lane == 0) atomicAdd(&cnt[tg], 1.0f);
}

__global__ void k_norm_loop(const float* __restrict__ cnt, float* __restrict__ loop) {
    size_t i = (size_t)blockIdx.x * blockDim.x + threadIdx.x;
    if (i >= (size_t)N_NODES * EDGE_F) return;
    int node = (int)(i / EDGE_F);
    loop[i] /= fmaxf(cnt[node], 1.0f);
}

// xl = x@Wl+bl, xr = x@Wr+br  — wave computes a 16x16 tile via WMMA bf16
__global__ void k_node_linear(const float* __restrict__ x,
                              const __bf16* __restrict__ pkl, const float* __restrict__ bl,
                              const __bf16* __restrict__ pkr, const float* __restrict__ br,
                              float* __restrict__ xl, float* __restrict__ xr) {
    int lane = threadIdx.x & 31;
    int wv = blockIdx.x * (blockDim.x >> 5) + (threadIdx.x >> 5);
    int tile = wv >> 3;                   // 16-node tile
    int sub  = wv & 7;                    // 2 matrices x 4 col tiles
    int mat  = sub >> 2, t = sub & 3;
    if (tile * 16 >= N_NODES) return;
    const __bf16* pk = mat ? pkr : pkl;
    const float*  b  = mat ? br  : bl;
    float* dst = mat ? xr : xl;
    int h = lane >> 4, n = lane & 15;
    const float* arow = x + (size_t)(tile * 16 + n) * NODE_F;   // 16B-aligned rows
    v8f acc = {};
#pragma unroll 4
    for (int c = 0; c < 12; ++c) {        // full chunks: k < 384
        v16bf a  = load_a_f4(arow, c * 32, h);
        v16bf bm = load_b_pk(pk, c, t, lane);
        acc = wmma_bf16(a, bm, acc);
    }
    {                                     // tail chunk: k = 384..387 valid of 384..415
        v16bf a  = load_a_tail(arow, 384, h, NODE_F);
        v16bf bm = load_b_pk(pk, 12, t, lane);
        acc = wmma_bf16(a, bm, acc);
    }
#pragma unroll
    for (int v = 0; v < 8; ++v) {
        int r = v + 8 * h;
        dst[(size_t)(tile * 16 + r) * HC + t * 16 + n] = acc[v] + b[t * 16 + n];
    }
}

// fused: ee = ea@We (WMMA), m = leaky(xl[src]+xr[tgt]+ee), score = m . att,
// store scores, atomic segment-max
__global__ void k_edge_score(const int* __restrict__ ei,
                             const float* __restrict__ eattr, const float* __restrict__ loop,
                             const __bf16* __restrict__ pke,
                             const float* __restrict__ xl, const float* __restrict__ xr,
                             const float* __restrict__ att,
                             float* __restrict__ score, unsigned* __restrict__ smax) {
    int lane = threadIdx.x & 31;
    int wv = blockIdx.x * (blockDim.x >> 5) + (threadIdx.x >> 5);
    if (wv >= E2 / 16) return;
    int base = wv * 16;
    int h = lane >> 4, n = lane & 15;
    int myEdge = base + n;                // A-fragment row for this lane
    const float* arow = (myEdge < NEDGE) ? (eattr + (size_t)myEdge * EDGE_F)
                                         : (loop + (size_t)(myEdge - NEDGE) * EDGE_F);
    v8f acc0 = {}, acc1 = {}, acc2 = {}, acc3 = {};
#pragma unroll 2
    for (int c = 0; c < 12; ++c) {        // full chunks: k < 384
        v16bf a  = load_a_s(arow, c * 32, h);
        acc0 = wmma_bf16(a, load_b_pk(pke, c, 0, lane), acc0);
        acc1 = wmma_bf16(a, load_b_pk(pke, c, 1, lane), acc1);
        acc2 = wmma_bf16(a, load_b_pk(pke, c, 2, lane), acc2);
        acc3 = wmma_bf16(a, load_b_pk(pke, c, 3, lane), acc3);
    }
    {                                     // tail: only k==384 valid
        v16bf a = load_a_tail(arow, 384, h, EDGE_F);
        acc0 = wmma_bf16(a, load_b_pk(pke, 12, 0, lane), acc0);
        acc1 = wmma_bf16(a, load_b_pk(pke, 12, 1, lane), acc1);
        acc2 = wmma_bf16(a, load_b_pk(pke, 12, 2, lane), acc2);
        acc3 = wmma_bf16(a, load_b_pk(pke, 12, 3, lane), acc3);
    }
    float s0[8], s1[8];
#pragma unroll
    for (int v = 0; v < 8; ++v) {
        int edge = base + v + 8 * h;
        int s = aug_src(ei, edge), tg = aug_tgt(ei, edge);
        const float* xls = xl + (size_t)s * HC;
        const float* xrt = xr + (size_t)tg * HC;
        float m0 = acc0[v] + xls[ 0 + n] + xrt[ 0 + n];
        float m1 = acc1[v] + xls[16 + n] + xrt[16 + n];
        float m2 = acc2[v] + xls[32 + n] + xrt[32 + n];
        float m3 = acc3[v] + xls[48 + n] + xrt[48 + n];
        m0 = m0 > 0.f ? m0 : NEG_SLOPE * m0;
        m1 = m1 > 0.f ? m1 : NEG_SLOPE * m1;
        m2 = m2 > 0.f ? m2 : NEG_SLOPE * m2;
        m3 = m3 > 0.f ? m3 : NEG_SLOPE * m3;
        s0[v] = m0 * att[ 0 + n] + m1 * att[16 + n];   // head 0
        s1[v] = m2 * att[32 + n] + m3 * att[48 + n];   // head 1
    }
#pragma unroll
    for (int off = 1; off < 16; off <<= 1) {
#pragma unroll
        for (int v = 0; v < 8; ++v) {
            s0[v] += __shfl_xor(s0[v], off, 32);
            s1[v] += __shfl_xor(s1[v], off, 32);
        }
    }
    if (n == 0) {
#pragma unroll
        for (int v = 0; v < 8; ++v) {
            int edge = base + v + 8 * h;
            int tg = aug_tgt(ei, edge);
            score[(size_t)edge * 2 + 0] = s0[v];
            score[(size_t)edge * 2 + 1] = s1[v];
            atomicMax(&smax[(size_t)tg * 2 + 0], f2ord(s0[v]));
            atomicMax(&smax[(size_t)tg * 2 + 1], f2ord(s1[v]));
        }
    }
}

// ex = exp(score - smax[tgt]); ssum[tgt] += ex  (score overwritten by ex)
__global__ void k_softmax_num(const int* __restrict__ ei, float* __restrict__ score,
                              const unsigned* __restrict__ smax, float* __restrict__ ssum) {
    size_t i = (size_t)blockIdx.x * blockDim.x + threadIdx.x;
    if (i >= (size_t)E2 * 2) return;
    int e = (int)(i >> 1), hh = (int)(i & 1);
    int tg = aug_tgt(ei, e);
    float ex = __expf(score[i] - ord2f(smax[(size_t)tg * 2 + hh]));
    score[i] = ex;
    atomicAdd(&ssum[(size_t)tg * 2 + hh], ex);
}

// hacc[tgt] += alpha * xl[src]  (one wave per augmented edge; lanes = 2 cols each)
__global__ void k_aggregate(const int* __restrict__ ei, const float* __restrict__ exv,
                            const float* __restrict__ ssum, const float* __restrict__ xl,
                            float* __restrict__ hacc) {
    int lane = threadIdx.x & 31;
    int wv = blockIdx.x * (blockDim.x >> 5) + (threadIdx.x >> 5);
    if (wv >= E2) return;
    int s = aug_src(ei, wv), tg = aug_tgt(ei, wv);
    float a0 = exv[(size_t)wv * 2 + 0] / (ssum[(size_t)tg * 2 + 0] + 1e-16f);
    float a1 = exv[(size_t)wv * 2 + 1] / (ssum[(size_t)tg * 2 + 1] + 1e-16f);
    atomicAdd(&hacc[(size_t)tg * HC + lane],      a0 * xl[(size_t)s * HC + lane]);
    atomicAdd(&hacc[(size_t)tg * HC + lane + 32], a1 * xl[(size_t)s * HC + lane + 32]);
}

__global__ void k_relu_bias(float* __restrict__ hacc, const float* __restrict__ conv_b) {
    size_t i = (size_t)blockIdx.x * blockDim.x + threadIdx.x;
    if (i >= (size_t)N_NODES * HC) return;
    hacc[i] = fmaxf(hacc[i] + conv_b[i & (HC - 1)], 0.0f);
}

// fused edge MLP: [h[src]|h[tgt]|q] @ W1 (WMMA) -> relu -> W2 -> softmax -> trust
__global__ void k_edge_mlp(const int* __restrict__ ei, const float* __restrict__ eattr,
                           const float* __restrict__ hn,
                           const __bf16* __restrict__ pkm, const float* __restrict__ b1,
                           const float* __restrict__ w2, const float* __restrict__ b2,
                           float* __restrict__ out) {
    int lane = threadIdx.x & 31;
    int wv = blockIdx.x * (blockDim.x >> 5) + (threadIdx.x >> 5);
    if (wv >= NEDGE / 16) return;
    int base = wv * 16;
    int h = lane >> 4, n = lane & 15;
    int myEdge = base + n;
    const float* hs = hn + (size_t)ei[myEdge] * HC;          // 256B-aligned rows
    const float* ht = hn + (size_t)ei[NEDGE + myEdge] * HC;
    const float* q  = eattr + (size_t)myEdge * EDGE_F;
    v8f acc0 = {}, acc1 = {}, acc2 = {}, acc3 = {};
#pragma unroll
    for (int c = 0; c < 2; ++c) {         // k 0..63   <- h[src]
        v16bf a = load_a_f4(hs, c * 32, h);
        acc0 = wmma_bf16(a, load_b_pk(pkm, c, 0, lane), acc0);
        acc1 = wmma_bf16(a, load_b_pk(pkm, c, 1, lane), acc1);
        acc2 = wmma_bf16(a, load_b_pk(pkm, c, 2, lane), acc2);
        acc3 = wmma_bf16(a, load_b_pk(pkm, c, 3, lane), acc3);
    }
#pragma unroll
    for (int c = 2; c < 4; ++c) {         // k 64..127 <- h[tgt]
        v16bf a = load_a_f4(ht, (c - 2) * 32, h);
        acc0 = wmma_bf16(a, load_b_pk(pkm, c, 0, lane), acc0);
        acc1 = wmma_bf16(a, load_b_pk(pkm, c, 1, lane), acc1);
        acc2 = wmma_bf16(a, load_b_pk(pkm, c, 2, lane), acc2);
        acc3 = wmma_bf16(a, load_b_pk(pkm, c, 3, lane), acc3);
    }
#pragma unroll 2
    for (int c = 4; c < 16; ++c) {        // k 128..511 <- q (first 384 attrs)
        v16bf a = load_a_s(q, (c - 4) * 32, h);
        acc0 = wmma_bf16(a, load_b_pk(pkm, c, 0, lane), acc0);
        acc1 = wmma_bf16(a, load_b_pk(pkm, c, 1, lane), acc1);
        acc2 = wmma_bf16(a, load_b_pk(pkm, c, 2, lane), acc2);
        acc3 = wmma_bf16(a, load_b_pk(pkm, c, 3, lane), acc3);
    }
    float l0[8], l1[8], l2[8];
#pragma unroll
    for (int v = 0; v < 8; ++v) {
        float h0 = fmaxf(acc0[v] + b1[ 0 + n], 0.0f);
        float h1 = fmaxf(acc1[v] + b1[16 + n], 0.0f);
        float h2 = fmaxf(acc2[v] + b1[32 + n], 0.0f);
        float h3 = fmaxf(acc3[v] + b1[48 + n], 0.0f);
        l0[v] = h0 * w2[( 0 + n) * 3 + 0] + h1 * w2[(16 + n) * 3 + 0]
              + h2 * w2[(32 + n) * 3 + 0] + h3 * w2[(48 + n) * 3 + 0];
        l1[v] = h0 * w2[( 0 + n) * 3 + 1] + h1 * w2[(16 + n) * 3 + 1]
              + h2 * w2[(32 + n) * 3 + 1] + h3 * w2[(48 + n) * 3 + 1];
        l2[v] = h0 * w2[( 0 + n) * 3 + 2] + h1 * w2[(16 + n) * 3 + 2]
              + h2 * w2[(32 + n) * 3 + 2] + h3 * w2[(48 + n) * 3 + 2];
    }
#pragma unroll
    for (int off = 1; off < 16; off <<= 1) {
#pragma unroll
        for (int v = 0; v < 8; ++v) {
            l0[v] += __shfl_xor(l0[v], off, 32);
            l1[v] += __shfl_xor(l1[v], off, 32);
            l2[v] += __shfl_xor(l2[v], off, 32);
        }
    }
    if (n == 0) {
#pragma unroll
        for (int v = 0; v < 8; ++v) {
            float a0 = l0[v] + b2[0], a1 = l1[v] + b2[1], a2 = l2[v] + b2[2];
            float mx = fmaxf(a0, fmaxf(a1, a2));
            float e0 = __expf(a0 - mx), e1 = __expf(a1 - mx), e2 = __expf(a2 - mx);
            out[base + v + 8 * h] = (0.5f * e1 + 1.0f * e2) / (e0 + e1 + e2);
        }
    }
}

// ---------------- launch ----------------

extern "C" void kernel_launch(void* const* d_in, const int* in_sizes, int n_in,
                              void* d_out, int out_size, void* d_ws, size_t ws_size,
                              hipStream_t stream) {
    const float* x       = (const float*)d_in[0];
    const int*   ei      = (const int*)d_in[1];
    const float* eattr   = (const float*)d_in[2];
    const float* lin_l_w = (const float*)d_in[3];
    const float* lin_l_b = (const float*)d_in[4];
    const float* lin_r_w = (const float*)d_in[5];
    const float* lin_r_b = (const float*)d_in[6];
    const float* lin_e_w = (const float*)d_in[7];
    const float* att     = (const float*)d_in[8];
    const float* conv_b  = (const float*)d_in[9];
    const float* mlp1_w  = (const float*)d_in[10];
    const float* mlp1_b  = (const float*)d_in[11];
    const float* mlp2_w  = (const float*)d_in[12];
    const float* mlp2_b  = (const float*)d_in[13];
    float* out = (float*)d_out;

    const size_t N = N_NODES;
    float* ws   = (float*)d_ws;
    float* cnt  = ws;                                   // N
    float* loop = cnt + N;                              // N*EDGE_F
    float* hacc = loop + N * (size_t)EDGE_F;            // N*HC
    float* ssum = hacc + N * (size_t)HC;                // N*2
    unsigned* smax = (unsigned*)(ssum + N * 2);         // N*2 (ordered uint)
    float* xl   = (float*)(smax + N * 2);               // N*HC
    float* xr   = xl + N * (size_t)HC;                  // N*HC
    float* score= xr + N * (size_t)HC;                  // E2*2
    // bf16 weight-fragment packs (32B aligned: float offsets are multiples of 16)
    float* pbase = score + (size_t)E2 * 2;
    __bf16* pkl = (__bf16*)(pbase);                     // 13*2048 bf16 = 13312 fl
    __bf16* pkr = (__bf16*)(pbase + 13312);
    __bf16* pke = (__bf16*)(pbase + 2 * 13312);
    __bf16* pkm = (__bf16*)(pbase + 3 * 13312);         // 16*2048 bf16

    // weight packs (independent of everything else)
    k_pack_w<<<(13 * 2048 + 255) / 256, 256, 0, stream>>>(lin_l_w, NODE_F, 13, pkl);
    k_pack_w<<<(13 * 2048 + 255) / 256, 256, 0, stream>>>(lin_r_w, NODE_F, 13, pkr);
    k_pack_w<<<(13 * 2048 + 255) / 256, 256, 0, stream>>>(lin_e_w, EDGE_F, 13, pke);
    k_pack_w<<<(16 * 2048 + 255) / 256, 256, 0, stream>>>(mlp1_w, EDGE_IN, 16, pkm);

    size_t nz = N * (size_t)(1 + EDGE_F + HC + 2 + 2);  // cnt..smax contiguous
    k_zero<<<(unsigned)((nz + 255) / 256), 256, 0, stream>>>(ws, nz);

    k_scatter_attr<<<NEDGE / 8, 256, 0, stream>>>(ei, eattr, cnt, loop);

    size_t nnorm = N * (size_t)EDGE_F;
    k_norm_loop<<<(unsigned)((nnorm + 255) / 256), 256, 0, stream>>>(cnt, loop);

    k_node_linear<<<N_NODES / 16, 256, 0, stream>>>(x, pkl, lin_l_b, pkr, lin_r_b, xl, xr);

    k_edge_score<<<(E2 / 16 + 7) / 8, 256, 0, stream>>>(ei, eattr, loop, pke,
                                                        xl, xr, att, score, smax);

    k_softmax_num<<<(unsigned)(((size_t)E2 * 2 + 255) / 256), 256, 0, stream>>>(
        ei, score, smax, ssum);

    k_aggregate<<<(E2 + 7) / 8, 256, 0, stream>>>(ei, score, ssum, xl, hacc);

    size_t nh = N * (size_t)HC;
    k_relu_bias<<<(unsigned)((nh + 255) / 256), 256, 0, stream>>>(hacc, conv_b);

    k_edge_mlp<<<NEDGE / 16 / 8, 256, 0, stream>>>(ei, eattr, hacc, pkm, mlp1_b,
                                                   mlp2_w, mlp2_b, out);
}
